// Defense_PointNet_59949153517667
// MI455X (gfx1250) — compile-verified, matrix operand order checked by reading, not run on hardware
//
#include <hip/hip_runtime.h>
#include <hip/hip_bf16.h>
#include <math.h>

// ---------------------------------------------------------------------------
// PointNet++ (SA x3 + heads) forward for MI455X (gfx1250, wave32, WMMA).
// GEMM-heavy point_conv MLPs run on v_wmma_f32_16x16x32_f16 (f16 in, f32 acc).
// knn / fps are selection- & latency-bound -> VALU + LDS kernels; fps uses
// wave32 shuffle argmax to minimize barriers on its serial critical path.
// ---------------------------------------------------------------------------

typedef _Float16 half_t;
typedef __attribute__((ext_vector_type(16))) _Float16 v16h;
typedef __attribute__((ext_vector_type(8)))  _Float16 v8h;
typedef __attribute__((ext_vector_type(8)))  float    v8f;

#define BATCH 8
#define KNN 32

// ---------------------------------------------------------------------------
// knn within radius: per-thread query, register top-32 via replace-max.
// Block = 128 queries (all same batch since N % 128 == 0); candidates staged
// through LDS in 128-point tiles.
// ---------------------------------------------------------------------------
template<int N>
__global__ __launch_bounds__(128)
void knn_kernel(const float* __restrict__ pos, int* __restrict__ idx,
                int* __restrict__ valid, float r2)
{
  const int q = blockIdx.x * 128 + threadIdx.x;   // global query over B*N
  const int b = q / N;
  const int i = q % N;
  const float* pb = pos + (size_t)b * N * 3;
  const float qx = pb[i * 3 + 0], qy = pb[i * 3 + 1], qz = pb[i * 3 + 2];

  __shared__ float sx[128], sy[128], sz[128];

  float kd[KNN]; int ki[KNN];
#pragma unroll
  for (int s = 0; s < KNN; ++s) { kd[s] = 1e30f; ki[s] = 0; }
  float cmax = 1e30f; int cslot = 0;

  for (int t0 = 0; t0 < N; t0 += 128) {
    const int j = t0 + threadIdx.x;
    sx[threadIdx.x] = pb[j * 3 + 0];
    sy[threadIdx.x] = pb[j * 3 + 1];
    sz[threadIdx.x] = pb[j * 3 + 2];
    __syncthreads();
    for (int jj = 0; jj < 128; ++jj) {
      const int cj = t0 + jj;
      const float dx = qx - sx[jj], dy = qy - sy[jj], dz = qz - sz[jj];
      float d2 = dx * dx + dy * dy + dz * dz;
      if (cj == i) d2 = 1e30f;                      // exclude self
      if (d2 < cmax) {
#pragma unroll
        for (int s = 0; s < KNN; ++s)
          if (s == cslot) { kd[s] = d2; ki[s] = cj; }
        cmax = -1.0f;
#pragma unroll
        for (int s = 0; s < KNN; ++s)
          if (kd[s] > cmax) { cmax = kd[s]; cslot = s; }
      }
    }
    __syncthreads();
  }
  int* op = idx   + (size_t)q * KNN;
  int* vp = valid + (size_t)q * KNN;
#pragma unroll
  for (int s = 0; s < KNN; ++s) { op[s] = ki[s]; vp[s] = (kd[s] <= r2) ? 1 : 0; }
}

// ---------------------------------------------------------------------------
// Farthest point sampling: one 256-thread (8-wave) workgroup per batch item.
// Points & running min-dist live in registers (N/256 per thread). Per-step
// argmax: wave32 shuffle reduction (no barrier) + 8-entry LDS exchange ->
// only 2 __syncthreads per sequential step (serial critical path).
// ---------------------------------------------------------------------------
template<int N, int NOUT>
__global__ __launch_bounds__(256)
void fps_kernel(const float* __restrict__ pos, int* __restrict__ out)
{
  const int b = blockIdx.x;
  const int t = threadIdx.x;
  constexpr int RN = N / 256;
  const float* pb = pos + (size_t)b * N * 3;
  int* ob = out + (size_t)b * NOUT;

  float px[RN], py[RN], pz[RN], mind[RN];
  const float x0 = pb[0], y0 = pb[1], z0 = pb[2];
#pragma unroll
  for (int r = 0; r < RN; ++r) {
    const int p = t + r * 256;
    px[r] = pb[p * 3 + 0]; py[r] = pb[p * 3 + 1]; pz[r] = pb[p * 3 + 2];
    const float dx = px[r] - x0, dy = py[r] - y0, dz = pz[r] - z0;
    mind[r] = dx * dx + dy * dy + dz * dz;
  }
  __shared__ float wv[8];
  __shared__ int   wp[8];
  __shared__ float cx, cy, cz;
  if (t == 0) ob[0] = 0;

  for (int it = 1; it < NOUT; ++it) {
    // local scan over this thread's points
    float bv = -1.0f; int bp = 0;
#pragma unroll
    for (int r = 0; r < RN; ++r)
      if (mind[r] > bv) { bv = mind[r]; bp = t + r * 256; }
    // wave32 shuffle argmax (no barriers)
#pragma unroll
    for (int m = 16; m > 0; m >>= 1) {
      const float ov = __shfl_xor(bv, m, 32);
      const int   op = __shfl_xor(bp, m, 32);
      if (ov > bv || (ov == bv && op < bp)) { bv = ov; bp = op; }
    }
    if ((t & 31) == 0) { wv[t >> 5] = bv; wp[t >> 5] = bp; }
    __syncthreads();
    // every thread scans the 8 wave winners (cheap, avoids another barrier)
    float gv = wv[0]; int gi = wp[0];
#pragma unroll
    for (int wj = 1; wj < 8; ++wj)
      if (wv[wj] > gv || (wv[wj] == gv && wp[wj] < gi)) { gv = wv[wj]; gi = wp[wj]; }
    // owner thread publishes the chosen point's coordinates
    if (t == (gi & 255)) {
      const int slot = gi >> 8;
#pragma unroll
      for (int r = 0; r < RN; ++r)
        if (r == slot) { cx = px[r]; cy = py[r]; cz = pz[r]; }
      ob[it] = gi;
    }
    __syncthreads();
    const float ax = cx, ay = cy, az = cz;
#pragma unroll
    for (int r = 0; r < RN; ++r) {
      const float dx = px[r] - ax, dy = py[r] - ay, dz = pz[r] - az;
      mind[r] = fminf(mind[r], dx * dx + dy * dy + dz * dz);
    }
    // no trailing barrier needed: wv/wp rewrites in the next step are gated by
    // the __syncthreads after them, and cx is rewritten only after that barrier
  }
}

// gather rows of C floats by per-batch fps indices
__global__ __launch_bounds__(256)
void gather_rows(const float* __restrict__ src, const int* __restrict__ fi,
                 float* __restrict__ dst, int Nin, int Nout, int C, int total)
{
  const int tid = blockIdx.x * 256 + threadIdx.x;
  if (tid >= total) return;
  const int c = tid % C;
  const int m = (tid / C) % Nout;
  const int b = tid / (C * Nout);
  const int j = fi[b * Nout + m];
  dst[tid] = src[((size_t)b * Nin + j) * C + c];
}

// ---------------------------------------------------------------------------
// Pack f32 weights [CIN x H] -> f16 WMMA B-fragment layout:
//   P[kt][nt][lane][16 halves], lane&15 = column, halves map K per the
//   16x16x32 B layout (K 0..7,16..23 lanes 0-15; K 8..15,24..31 lanes 16-31).
// Rows k >= CIN are zero (K padding).
// ---------------------------------------------------------------------------
__global__ __launch_bounds__(256)
void pack_w(const float* __restrict__ W, half_t* __restrict__ P,
            int CIN, int CINP, int H)
{
  const int tid = blockIdx.x * 256 + threadIdx.x;
  const int total = (CINP / 32) * (H / 16) * 32;
  if (tid >= total) return;
  const int lane = tid & 31;
  const int nt   = (tid >> 5) % (H / 16);
  const int kt   = tid / (32 * (H / 16));
  const int col  = nt * 16 + (lane & 15);
  half_t* o = P + (size_t)tid * 16;
#pragma unroll
  for (int h = 0; h < 16; ++h) {
    const int k = kt * 32 + ((h < 8) ? h : h + 8) + ((lane >= 16) ? 8 : 0);
    const float v = (k < CIN) ? W[(size_t)k * H + col] : 0.0f;
    o[h] = (half_t)v;
  }
}

// ---------------------------------------------------------------------------
// Build gathered neighbor features (one batch): row (i*K+k) =
// [x[idx[i,k]] (CX), pos[idx]-pos[i] (3), zero pad to CINP], f16 row-major.
// ---------------------------------------------------------------------------
__global__ __launch_bounds__(256)
void build_feat(const float* __restrict__ pos, const float* __restrict__ x,
                const int* __restrict__ idx, half_t* __restrict__ A,
                int Nl, int CX, int CINP)
{
  const int r = blockIdx.x * 256 + threadIdx.x;
  if (r >= Nl * KNN) return;
  const int i = r >> 5;
  const int j = idx[r];
  half_t* o = A + (size_t)r * CINP;
  int c = 0;
  for (; c < CX; ++c) o[c] = (half_t)x[(size_t)j * CX + c];
  o[c++] = (half_t)(pos[j * 3 + 0] - pos[i * 3 + 0]);
  o[c++] = (half_t)(pos[j * 3 + 1] - pos[i * 3 + 1]);
  o[c++] = (half_t)(pos[j * 3 + 2] - pos[i * 3 + 2]);
  for (; c < CINP; ++c) o[c] = (half_t)0.0f;
}

// ---------------------------------------------------------------------------
// WMMA GEMM: Out[M x HOUT] = op(A[M x CINP] * Wp + bias), f16 A, packed f16 B,
// f32 accumulate. 4 waves/block, each wave: one 16-row tile x CT col tiles.
// A fragment: lanes 0-15 row=lane&15 K 0..7/16..23; lanes 16-31 K 8..15/24..31
// (two b128 loads). B fragment: one contiguous v16h load from packed layout.
// D layout: VGPR v, lanes 0-15 -> (M=v, N=lane), lanes 16-31 -> (M=v+8).
// ---------------------------------------------------------------------------
template<int CINP, int HOUT, bool RELU, bool OUTF32>
__global__ __launch_bounds__(128)
void wmma_gemm(const half_t* __restrict__ A, const half_t* __restrict__ Wp,
               const float* __restrict__ bias, void* __restrict__ Out)
{
  constexpr int NCT = HOUT / 16;
  constexpr int CT  = (NCT > 8) ? 8 : NCT;
  const int lane = threadIdx.x & 31;
  const int wave = threadIdx.x >> 5;
  const int mt   = blockIdx.x * 4 + wave;
  const int nt0  = blockIdx.y * CT;
  const int mbase = mt * 16;
  const int arow  = mbase + (lane & 15);
  const int asel  = (lane >> 4) * 8;           // half-offset within K-group

  v8f acc[CT];
  const v8f zero = {0.f, 0.f, 0.f, 0.f, 0.f, 0.f, 0.f, 0.f};
#pragma unroll
  for (int n = 0; n < CT; ++n) acc[n] = zero;

#pragma unroll
  for (int kt = 0; kt < CINP / 32; ++kt) {
    const half_t* ap = A + (size_t)arow * CINP + kt * 32 + asel;
    const v8h a0 = *reinterpret_cast<const v8h*>(ap);
    const v8h a1 = *reinterpret_cast<const v8h*>(ap + 16);
    const v16h afrag = __builtin_shufflevector(a0, a1,
        0, 1, 2, 3, 4, 5, 6, 7, 8, 9, 10, 11, 12, 13, 14, 15);
#pragma unroll
    for (int n = 0; n < CT; ++n) {
      const v16h bfrag = *reinterpret_cast<const v16h*>(
          Wp + ((size_t)(kt * NCT + nt0 + n) * 32 + lane) * 16);
      acc[n] = __builtin_amdgcn_wmma_f32_16x16x32_f16(
          false, afrag, false, bfrag, (short)0, acc[n], false, false);
    }
  }

#pragma unroll
  for (int n = 0; n < CT; ++n) {
    const int col  = (nt0 + n) * 16 + (lane & 15);
    const float bv = bias[col];
    const int mofs = (lane < 16) ? 0 : 8;
#pragma unroll
    for (int v = 0; v < 8; ++v) {
      float val = acc[n][v] + bv;
      if (RELU) val = fmaxf(val, 0.0f);
      const size_t off = (size_t)(mbase + mofs + v) * HOUT + col;
      if constexpr (OUTF32) reinterpret_cast<float*>(Out)[off] = val;
      else                  reinterpret_cast<half_t*>(Out)[off] = (half_t)val;
    }
  }
}

// masked max over K neighbors + trailing ReLU (x = relu(point_conv(...)))
__global__ __launch_bounds__(256)
void mask_max_relu(const float* __restrict__ G, const int* __restrict__ valid,
                   float* __restrict__ X, int Nl, int C)
{
  const int tid = blockIdx.x * 256 + threadIdx.x;
  if (tid >= Nl * C) return;
  const int c = tid % C;
  const int i = tid / C;
  const float* g  = G + (size_t)i * KNN * C + c;
  const int*   vp = valid + (size_t)i * KNN;
  float m = -1e9f;
  for (int k = 0; k < KNN; ++k) {
    const float h = g[(size_t)k * C];
    m = fmaxf(m, vp[k] ? h : -1e9f);
  }
  X[tid] = fmaxf(m, 0.0f);
}

// global max pool [B,512,256] -> [B,256]
__global__ __launch_bounds__(256)
void gmax_kernel(const float* __restrict__ x3, float* __restrict__ feat)
{
  const int b = blockIdx.x, c = threadIdx.x;
  float m = -1e30f;
  for (int i = 0; i < 512; ++i)
    m = fmaxf(m, x3[((size_t)b * 512 + i) * 256 + c]);
  feat[b * 256 + c] = m;
}

// classifier + defense heads, one block per batch element (tiny)
__global__ __launch_bounds__(256)
void head_kernel(const float* __restrict__ feat,
                 const float* Wc1, const float* bc1, const float* Wc2, const float* bc2,
                 const float* Wc3, const float* bc3, const float* Wd1, const float* bd1,
                 const float* Wd2, const float* bd2, float* __restrict__ out)
{
  const int b = blockIdx.x, t = threadIdx.x;
  __shared__ float f[256], h1[256], h2[256], hd[256], lg[40], dd[2];
  f[t] = feat[b * 256 + t];
  __syncthreads();
  float a = bc1[t];
  for (int k = 0; k < 256; ++k) a += f[k] * Wc1[k * 256 + t];
  h1[t] = fmaxf(a, 0.0f);
  __syncthreads();
  a = bc2[t];
  for (int k = 0; k < 256; ++k) a += h1[k] * Wc2[k * 256 + t];
  h2[t] = fmaxf(a, 0.0f);
  __syncthreads();
  if (t < 40) {
    a = bc3[t];
    for (int k = 0; k < 256; ++k) a += h2[k] * Wc3[k * 40 + t];
    lg[t] = a;
  }
  float ad = bd1[t];
  for (int k = 0; k < 256; ++k) ad += f[k] * Wd1[k * 256 + t];
  hd[t] = fmaxf(ad, 0.0f);
  __syncthreads();
  if (t < 2) {
    a = bd2[t];
    for (int k = 0; k < 256; ++k) a += hd[k] * Wd2[k * 2 + t];
    dd[t] = a;
  }
  __syncthreads();
  if (t == 0) {
    float m = -1e30f;
    for (int j = 0; j < 40; ++j) m = fmaxf(m, lg[j]);
    float s = 0.0f;
    for (int j = 0; j < 40; ++j) s += expf(lg[j] - m);
    const float ls = logf(s);
    for (int j = 0; j < 40; ++j) out[b * 40 + j] = lg[j] - m - ls;
    const float m2 = fmaxf(dd[0], dd[1]);
    const float s2 = expf(dd[0] - m2) + expf(dd[1] - m2);
    const float l2 = logf(s2);
    out[BATCH * 40 + b * 2 + 0] = dd[0] - m2 - l2;
    out[BATCH * 40 + b * 2 + 1] = dd[1] - m2 - l2;
  }
}

// ---------------------------------------------------------------------------
extern "C" void kernel_launch(void* const* d_in, const int* in_sizes, int n_in,
                              void* d_out, int out_size, void* d_ws, size_t ws_size,
                              hipStream_t stream)
{
  const float* pos = (const float*)d_in[0];
  const float* W1a = (const float*)d_in[1];  const float* b1a = (const float*)d_in[2];
  const float* W1b = (const float*)d_in[3];  const float* b1b = (const float*)d_in[4];
  const float* W2a = (const float*)d_in[5];  const float* b2a = (const float*)d_in[6];
  const float* W2b = (const float*)d_in[7];  const float* b2b = (const float*)d_in[8];
  const float* W3a = (const float*)d_in[9];  const float* b3a = (const float*)d_in[10];
  const float* W3b = (const float*)d_in[11]; const float* b3b = (const float*)d_in[12];
  const float* Wc1 = (const float*)d_in[13]; const float* bc1 = (const float*)d_in[14];
  const float* Wc2 = (const float*)d_in[15]; const float* bc2 = (const float*)d_in[16];
  const float* Wc3 = (const float*)d_in[17]; const float* bc3 = (const float*)d_in[18];
  const float* Wd1 = (const float*)d_in[19]; const float* bd1 = (const float*)d_in[20];
  const float* Wd2 = (const float*)d_in[21]; const float* bd2 = (const float*)d_in[22];
  float* out = (float*)d_out;

  // ---- workspace carve ----
  char* w = (char*)d_ws;
  size_t off = 0;
  auto alloc = [&](size_t bytes) -> void* {
    void* p = w + off;
    off = (off + bytes + 255) & ~(size_t)255;
    return p;
  };
  int*    idx   = (int*)   alloc((size_t)BATCH * 4096 * KNN * 4);
  int*    valid = (int*)   alloc((size_t)BATCH * 4096 * KNN * 4);
  int*    fi    = (int*)   alloc((size_t)BATCH * 2048 * 4);
  float*  pos2  = (float*) alloc((size_t)BATCH * 2048 * 3 * 4);
  float*  pos3  = (float*) alloc((size_t)BATCH * 512 * 3 * 4);
  float*  x1    = (float*) alloc((size_t)BATCH * 4096 * 64 * 4);
  float*  x2g   = (float*) alloc((size_t)BATCH * 2048 * 64 * 4);
  float*  x2    = (float*) alloc((size_t)BATCH * 2048 * 128 * 4);
  float*  x3g   = (float*) alloc((size_t)BATCH * 512 * 128 * 4);
  float*  x3    = (float*) alloc((size_t)BATCH * 512 * 256 * 4);
  float*  feat  = (float*) alloc((size_t)BATCH * 256 * 4);
  half_t* Wp1a  = (half_t*)alloc((size_t)32  * 64  * 2);
  half_t* Wp1b  = (half_t*)alloc((size_t)64  * 64  * 2);
  half_t* Wp2a  = (half_t*)alloc((size_t)96  * 128 * 2);
  half_t* Wp2b  = (half_t*)alloc((size_t)128 * 128 * 2);
  half_t* Wp3a  = (half_t*)alloc((size_t)160 * 256 * 2);
  half_t* Wp3b  = (half_t*)alloc((size_t)256 * 256 * 2);
  half_t* AF    = (half_t*)alloc((size_t)65536 * 96 * 2);    // per-batch feat (max level2)
  half_t* HB    = (half_t*)alloc((size_t)131072 * 64 * 2);   // per-batch hidden f16
  float*  G2    = (float*) alloc((size_t)131072 * 64 * 4);   // per-batch gemm2 f32
  (void)ws_size; (void)n_in; (void)in_sizes; (void)out_size;

  // ---- pack weights into WMMA B-fragment layout ----
  auto pk = [&](const float* W, half_t* P, int cin, int cinp, int h) {
    const int total = (cinp / 32) * (h / 16) * 32;
    pack_w<<<(total + 255) / 256, 256, 0, stream>>>(W, P, cin, cinp, h);
  };
  pk(W1a, Wp1a, 3,   32,  64);  pk(W1b, Wp1b, 64,  64,  64);
  pk(W2a, Wp2a, 67,  96,  128); pk(W2b, Wp2b, 128, 128, 128);
  pk(W3a, Wp3a, 131, 160, 256); pk(W3b, Wp3b, 256, 256, 256);

  // ---- Level 1: N=4096, r=0.2, feat rel(3)->pad32 -> 64 -> 64 ----
  knn_kernel<4096><<<BATCH * 4096 / 128, 128, 0, stream>>>(pos, idx, valid, 0.04f);
  for (int b = 0; b < BATCH; ++b) {
    build_feat<<<(4096 * KNN) / 256, 256, 0, stream>>>(
        pos + (size_t)b * 4096 * 3, nullptr, idx + (size_t)b * 4096 * KNN, AF, 4096, 0, 32);
    wmma_gemm<32, 64, true, false><<<dim3(131072 / 64, 1), 128, 0, stream>>>(AF, Wp1a, b1a, HB);
    wmma_gemm<64, 64, false, true><<<dim3(131072 / 64, 1), 128, 0, stream>>>(HB, Wp1b, b1b, G2);
    mask_max_relu<<<(4096 * 64) / 256, 256, 0, stream>>>(
        G2, valid + (size_t)b * 4096 * KNN, x1 + (size_t)b * 4096 * 64, 4096, 64);
  }

  // ---- FPS 4096 -> 2048, gather ----
  fps_kernel<4096, 2048><<<BATCH, 256, 0, stream>>>(pos, fi);
  gather_rows<<<(BATCH * 2048 * 3 + 255) / 256, 256, 0, stream>>>(pos, fi, pos2, 4096, 2048, 3, BATCH * 2048 * 3);
  gather_rows<<<(BATCH * 2048 * 64 + 255) / 256, 256, 0, stream>>>(x1, fi, x2g, 4096, 2048, 64, BATCH * 2048 * 64);

  // ---- Level 2: N=2048, r=0.4, feat 67->pad96 -> 128 -> 128 ----
  knn_kernel<2048><<<BATCH * 2048 / 128, 128, 0, stream>>>(pos2, idx, valid, 0.16f);
  for (int b = 0; b < BATCH; ++b) {
    build_feat<<<(2048 * KNN) / 256, 256, 0, stream>>>(
        pos2 + (size_t)b * 2048 * 3, x2g + (size_t)b * 2048 * 64,
        idx + (size_t)b * 2048 * KNN, AF, 2048, 64, 96);
    wmma_gemm<96, 128, true, false><<<dim3(65536 / 64, 1), 128, 0, stream>>>(AF, Wp2a, b2a, HB);
    wmma_gemm<128, 128, false, true><<<dim3(65536 / 64, 1), 128, 0, stream>>>(HB, Wp2b, b2b, G2);
    mask_max_relu<<<(2048 * 128) / 256, 256, 0, stream>>>(
        G2, valid + (size_t)b * 2048 * KNN, x2 + (size_t)b * 2048 * 128, 2048, 128);
  }

  // ---- FPS 2048 -> 512, gather ----
  fps_kernel<2048, 512><<<BATCH, 256, 0, stream>>>(pos2, fi);
  gather_rows<<<(BATCH * 512 * 3 + 255) / 256, 256, 0, stream>>>(pos2, fi, pos3, 2048, 512, 3, BATCH * 512 * 3);
  gather_rows<<<(BATCH * 512 * 128 + 255) / 256, 256, 0, stream>>>(x2, fi, x3g, 2048, 512, 128, BATCH * 512 * 128);

  // ---- Level 3: N=512, r=1.0, feat 131->pad160 -> 256 -> 256 ----
  knn_kernel<512><<<BATCH * 512 / 128, 128, 0, stream>>>(pos3, idx, valid, 1.0f);
  for (int b = 0; b < BATCH; ++b) {
    build_feat<<<(512 * KNN) / 256, 256, 0, stream>>>(
        pos3 + (size_t)b * 512 * 3, x3g + (size_t)b * 512 * 128,
        idx + (size_t)b * 512 * KNN, AF, 512, 128, 160);
    wmma_gemm<160, 256, true, false><<<dim3(16384 / 64, 2), 128, 0, stream>>>(AF, Wp3a, b3a, HB);
    wmma_gemm<256, 256, false, true><<<dim3(16384 / 64, 2), 128, 0, stream>>>(HB, Wp3b, b3b, G2);
    mask_max_relu<<<(512 * 256) / 256, 256, 0, stream>>>(
        G2, valid + (size_t)b * 512 * KNN, x3 + (size_t)b * 512 * 256, 512, 256);
  }

  // ---- global max pool + heads ----
  gmax_kernel<<<BATCH, 256, 0, stream>>>(x3, feat);
  head_kernel<<<BATCH, 256, 0, stream>>>(feat, Wc1, bc1, Wc2, bc2, Wc3, bc3,
                                         Wd1, bd1, Wd2, bd2, out);
}
